// VectorQuantizer_29970281791598
// MI455X (gfx1250) — matrix operand order repaired and checked by reading, not compiled, not run on hardware
//
#include <hip/hip_runtime.h>
#include <stdint.h>

// Problem constants (from reference)
#define NN 8192
#define DD 2048
#define KK 10000
#define ROWS_PER_BLOCK 32          // two 16-row WMMA tiles share each B fragment

typedef __attribute__((ext_vector_type(16))) __bf16 v16bf;
typedef __attribute__((ext_vector_type(8)))  float  v8f;

union Frag  { uint4 q[2]; v16bf v; };          // 32 bytes = 16 bf16
union Pack8 { unsigned short h[8]; uint4 q; }; // 8 bf16 -> one b128 store

__device__ __forceinline__ unsigned short f2bf(float f) {
  unsigned int b = __float_as_uint(f);
  b += 0x7FFFu + ((b >> 16) & 1u);             // round-to-nearest-even
  return (unsigned short)(b >> 16);
}
// monotonic float -> uint mapping for max-reduction
__device__ __forceinline__ unsigned int ordf(float f) {
  int b = __float_as_int(f);
  return (b < 0) ? ~(unsigned int)b : ((unsigned int)b | 0x80000000u);
}

// Load one A fragment (16x32 bf16 tile slice) from LDS.
// Lanes 0-15: K=[kb..kb+7],[kb+16..kb+23] with kb=d0; lanes 16-31: kb=d0+8.
__device__ __forceinline__ Frag loadA1(const unsigned short* aRow, int kb) {
  Frag a;
  a.q[0] = *(const uint4*)(aRow + kb);
  a.q[1] = *(const uint4*)(aRow + kb + 16);
  return a;
}
// Load 4 B fragments (32x16 bf16) from global; dd already includes half*8.
__device__ __forceinline__ void loadB4(Frag* b,
                                       const unsigned short* const* brow, int dd) {
#pragma unroll
  for (int t = 0; t < 4; ++t) {
    b[t].q[0] = *(const uint4*)(brow[t] + dd);
    b[t].q[1] = *(const uint4*)(brow[t] + dd + 16);
  }
}

// ---------------------------------------------------------------------------
// Kernel 1: L2-normalize rows of x (NN rows) and weight (KK rows) -> bf16
// ---------------------------------------------------------------------------
__global__ void __launch_bounds__(256) vq_normalize(
    const float* __restrict__ x, const float* __restrict__ w,
    unsigned short* __restrict__ xn, unsigned short* __restrict__ wn) {
  int row = blockIdx.x;
  const float* src;
  unsigned short* dst;
  if (row < NN) { src = x + (size_t)row * DD;        dst = xn + (size_t)row * DD; }
  else          { src = w + (size_t)(row - NN) * DD; dst = wn + (size_t)(row - NN) * DD; }

  int t = threadIdx.x;
  const float4* s4 = (const float4*)src;
  float4 a = s4[t * 2 + 0];
  float4 b = s4[t * 2 + 1];
  float ss = a.x*a.x + a.y*a.y + a.z*a.z + a.w*a.w
           + b.x*b.x + b.y*b.y + b.z*b.z + b.w*b.w;
  for (int off = 16; off > 0; off >>= 1) ss += __shfl_down(ss, off, 32);

  __shared__ float warpS[8];
  __shared__ float scaleSh;
  int lane = t & 31, wv = t >> 5;
  if (lane == 0) warpS[wv] = ss;
  __syncthreads();
  if (t == 0) {
    float tot = 0.f;
    for (int i = 0; i < 8; ++i) tot += warpS[i];
    scaleSh = 1.0f / fmaxf(sqrtf(tot), 1e-12f);   // matches v / max(||v||, eps)
  }
  __syncthreads();
  float sc = scaleSh;

  Pack8 p;
  p.h[0] = f2bf(a.x * sc); p.h[1] = f2bf(a.y * sc);
  p.h[2] = f2bf(a.z * sc); p.h[3] = f2bf(a.w * sc);
  p.h[4] = f2bf(b.x * sc); p.h[5] = f2bf(b.y * sc);
  p.h[6] = f2bf(b.z * sc); p.h[7] = f2bf(b.w * sc);
  ((uint4*)dst)[t] = p.q;
}

// ---------------------------------------------------------------------------
// Kernel 2: 32 rows per block (2 A-tiles staged in 128 KB dynamic LDS),
// 8 waves each cover KK/8 = 1250 codes. Per 64-code chunk: d-loop with
// double-buffered B fragments; 8 independent v_wmma per load group.
// ---------------------------------------------------------------------------
__global__ void __launch_bounds__(256, 2) vq_argmax(
    const unsigned short* __restrict__ xn,
    const unsigned short* __restrict__ wn,
    unsigned long long* __restrict__ gBest) {
  extern __shared__ __align__(16) unsigned char smemRaw[];
  unsigned short* aTile = (unsigned short*)smemRaw;   // 32 x DD bf16 = 128 KB

  int tid = threadIdx.x;

  // Stage the 32 x D bf16 A-slab into LDS (reused by every chunk).
  {
    const uint4* s = (const uint4*)(xn + (size_t)blockIdx.x * ROWS_PER_BLOCK * DD);
    uint4* d = (uint4*)aTile;
    for (int i = tid; i < (ROWS_PER_BLOCK * DD) / 8; i += 256) d[i] = s[i];
  }
  __syncthreads();

  int wave = tid >> 5, lane = tid & 31;
  int half = lane >> 4, r15 = lane & 15;
  int kstart = wave * (KK / 8);
  int kend   = kstart + (KK / 8);

  float bestv[2][8];
  unsigned int besti[2][8];
#pragma unroll
  for (int g = 0; g < 2; ++g)
#pragma unroll
    for (int r = 0; r < 8; ++r) { bestv[g][r] = -3.4e38f; besti[g][r] = 0u; }

  const unsigned short* aRow0 = aTile + (0 * 16 + r15) * DD;
  const unsigned short* aRow1 = aTile + (1 * 16 + r15) * DD;

  for (int jc = kstart; jc < kend; jc += 64) {
    const unsigned short* brow[4];
#pragma unroll
    for (int t = 0; t < 4; ++t) {
      int j = jc + 16 * t + r15;                 // B lane -> code column (L&15)
      if (j > KK - 1) j = KK - 1;                // clamp (masked later)
      brow[t] = wn + (size_t)j * DD;
    }
    // Warm L2/WGP$ for the next chunk's rows (global_prefetch_b8).
    if (jc + 64 < kend) {
#pragma unroll
      for (int t = 0; t < 4; ++t) {
        int j = jc + 64 + 16 * t + r15;
        if (j > KK - 1) j = KK - 1;
        __builtin_prefetch(wn + (size_t)j * DD, 0, 1);
      }
    }

    v8f acc[2][4];
#pragma unroll
    for (int g = 0; g < 2; ++g)
#pragma unroll
      for (int t = 0; t < 4; ++t)
#pragma unroll
        for (int r = 0; r < 8; ++r) acc[g][t][r] = 0.0f;

    int hb16 = half * 8;                         // B K-offset per half-wave
    Frag bCur[4], bNxt[4];
    loadB4(bCur, brow, 0 + hb16);                // prologue: d0 = 0

#pragma unroll 1
    for (int d0 = 0; d0 < DD; d0 += 64) {
      // --- group 0: consume bCur (d0), fetch bNxt (d0+32) ---
      loadB4(bNxt, brow, (d0 + 32) + hb16);
      {
        int kb = d0 + half * 8;
        Frag a0 = loadA1(aRow0, kb);
        Frag a1 = loadA1(aRow1, kb);
#pragma unroll
        for (int t = 0; t < 4; ++t) {
          acc[0][t] = __builtin_amdgcn_wmma_f32_16x16x32_bf16(
              false, a0.v, false, bCur[t].v, (short)0, acc[0][t], false, false);
          acc[1][t] = __builtin_amdgcn_wmma_f32_16x16x32_bf16(
              false, a1.v, false, bCur[t].v, (short)0, acc[1][t], false, false);
        }
      }
      // --- group 1: consume bNxt (d0+32), fetch bCur (d0+64, clamped) ---
      int dn = d0 + 64;
      if (dn >= DD) dn = DD - 64;                // redundant (cached) reload at end
      loadB4(bCur, brow, dn + hb16);
      {
        int kb = (d0 + 32) + half * 8;
        Frag a0 = loadA1(aRow0, kb);
        Frag a1 = loadA1(aRow1, kb);
#pragma unroll
        for (int t = 0; t < 4; ++t) {
          acc[0][t] = __builtin_amdgcn_wmma_f32_16x16x32_bf16(
              false, a0.v, false, bNxt[t].v, (short)0, acc[0][t], false, false);
          acc[1][t] = __builtin_amdgcn_wmma_f32_16x16x32_bf16(
              false, a1.v, false, bNxt[t].v, (short)0, acc[1][t], false, false);
        }
      }
    }

    // Per-lane best update: C/D VGPR r holds row (g*16 + r + 8*half), col = L&15
#pragma unroll
    for (int t = 0; t < 4; ++t) {
      int j = jc + 16 * t + r15;
      if (j < kend) {
#pragma unroll
        for (int g = 0; g < 2; ++g)
#pragma unroll
          for (int r = 0; r < 8; ++r) {
            float v = acc[g][t][r];
            if (v > bestv[g][r]) { bestv[g][r] = v; besti[g][r] = (unsigned)j; }
          }
      }
    }
  }

  // Reuse A-tile LDS as the 32-row best table (A no longer needed).
  __syncthreads();
  unsigned long long* bestLds = (unsigned long long*)aTile;
  if (tid < ROWS_PER_BLOCK) bestLds[tid] = 0ull;
  __syncthreads();

  // 16-lane butterfly max on packed (ordered-val << 32) | ~idx  (ties -> low idx)
#pragma unroll
  for (int g = 0; g < 2; ++g)
#pragma unroll
    for (int r = 0; r < 8; ++r) {
      unsigned long long pk =
          ((unsigned long long)ordf(bestv[g][r]) << 32) |
          (unsigned long long)(unsigned int)(~besti[g][r]);
      for (int s = 1; s < 16; s <<= 1) {
        unsigned int ohi = __shfl_xor((unsigned int)(pk >> 32), s, 16);
        unsigned int olo = __shfl_xor((unsigned int)pk,        s, 16);
        unsigned long long o = ((unsigned long long)ohi << 32) | olo;
        if (o > pk) pk = o;
      }
      if (r15 == 0) atomicMax(&bestLds[g * 16 + r + 8 * half], pk);
    }
  __syncthreads();
  if (tid < ROWS_PER_BLOCK) gBest[blockIdx.x * ROWS_PER_BLOCK + tid] = bestLds[tid];
}

// ---------------------------------------------------------------------------
// Kernel 3: gather weight[idx] -> quantized output; per-row squared error.
// ---------------------------------------------------------------------------
__global__ void __launch_bounds__(256) vq_finalize(
    const float* __restrict__ w, const float* __restrict__ x,
    const unsigned long long* __restrict__ gBest,
    float* __restrict__ out, float* __restrict__ partial) {
  int row = blockIdx.x, t = threadIdx.x;
  unsigned long long pk = gBest[row];
  unsigned int idx = ~(unsigned int)pk;
  if (idx >= KK) idx = KK - 1;

  const float4* wr = (const float4*)(w + (size_t)idx * DD);
  const float4* xr = (const float4*)(x + (size_t)row * DD);
  float4* orow = (float4*)(out + (size_t)row * DD);

  float ss = 0.f;
#pragma unroll
  for (int i = 0; i < 2; ++i) {
    float4 wv = wr[t * 2 + i];
    float4 xv = xr[t * 2 + i];
    orow[t * 2 + i] = wv;                       // STE output == weight[idx]
    float dx = wv.x - xv.x, dy = wv.y - xv.y;
    float dz = wv.z - xv.z, dw = wv.w - xv.w;
    ss += dx * dx + dy * dy + dz * dz + dw * dw;
  }
  for (int off = 16; off > 0; off >>= 1) ss += __shfl_down(ss, off, 32);

  __shared__ float warpS[8];
  int lane = t & 31, wv2 = t >> 5;
  if (lane == 0) warpS[wv2] = ss;
  __syncthreads();
  if (t == 0) {
    float tot = 0.f;
    for (int i = 0; i < 8; ++i) tot += warpS[i];
    partial[row] = tot;
    out[(size_t)NN * DD + 1 + row] = (float)idx;  // indices as float
  }
}

// ---------------------------------------------------------------------------
// Kernel 4: loss = (1 + 0.25) * mean((quantized - x)^2)
// ---------------------------------------------------------------------------
__global__ void __launch_bounds__(256) vq_loss(
    const float* __restrict__ partial, float* __restrict__ out) {
  int t = threadIdx.x;
  float s = 0.f;
  for (int i = t; i < NN; i += 256) s += partial[i];
  for (int off = 16; off > 0; off >>= 1) s += __shfl_down(s, off, 32);
  __shared__ float warpS[8];
  int lane = t & 31, wv = t >> 5;
  if (lane == 0) warpS[wv] = s;
  __syncthreads();
  if (t == 0) {
    float tot = 0.f;
    for (int i = 0; i < 8; ++i) tot += warpS[i];
    out[(size_t)NN * DD] = 1.25f * tot / (float)((size_t)NN * DD);
  }
}

// ---------------------------------------------------------------------------
extern "C" void kernel_launch(void* const* d_in, const int* in_sizes, int n_in,
                              void* d_out, int out_size, void* d_ws, size_t ws_size,
                              hipStream_t stream) {
  (void)in_sizes; (void)n_in; (void)out_size; (void)ws_size;
  const float* x = (const float*)d_in[0];
  const float* w = (const float*)d_in[1];

  // workspace layout: xn bf16 | wn bf16 | gBest u64[NN] | partial f32[NN]
  unsigned short* xn = (unsigned short*)d_ws;
  unsigned short* wn = xn + (size_t)NN * DD;
  size_t off = (((size_t)NN * DD + (size_t)KK * DD) * 2 + 255) & ~(size_t)255;
  unsigned long long* gBest = (unsigned long long*)((char*)d_ws + off);
  float* partial = (float*)((char*)d_ws + off + (size_t)NN * 8);
  float* out = (float*)d_out;

  size_t ldsBytes = (size_t)ROWS_PER_BLOCK * DD * sizeof(unsigned short); // 128 KB
  vq_normalize<<<NN + KK, 256, 0, stream>>>(x, w, xn, wn);
  vq_argmax  <<<NN / ROWS_PER_BLOCK, 256, ldsBytes, stream>>>(xn, wn, gBest);
  vq_finalize<<<NN, 256, 0, stream>>>(w, x, gBest, out, partial);
  vq_loss    <<<1, 256, 0, stream>>>(partial, out);
}
